// QuantGATLayer_75788992905524
// MI455X (gfx1250) — matrix operand order repaired and checked by reading, not compiled, not run on hardware
//
#include <hip/hip_runtime.h>
#include <math.h>

typedef __attribute__((ext_vector_type(16))) _Float16 v16h;
typedef __attribute__((ext_vector_type(8)))  float    v8f;

__device__ __forceinline__ void atomicMaxFloat(float* addr, float v) {
  // order-preserving int-punned float max (init must be -inf)
  if (v >= 0.f) atomicMax((int*)addr, __float_as_int(v));
  else          atomicMin((unsigned int*)addr, __float_as_uint(v));
}

// ---------- prep kernels ----------

__global__ void k_fill(float* __restrict__ p, float v, int n) {
  int i = blockIdx.x * blockDim.x + threadIdx.x;
  if (i < n) p[i] = v;
}

__global__ void k_absmax_w(const float* __restrict__ w, float* __restrict__ out, int n) {
  __shared__ float red[256];
  float m = 0.f;
  for (int i = threadIdx.x; i < n; i += 256) m = fmaxf(m, fabsf(w[i]));
  red[threadIdx.x] = m;
  __syncthreads();
  for (int s = 128; s > 0; s >>= 1) {
    if (threadIdx.x < s) red[threadIdx.x] = fmaxf(red[threadIdx.x], red[threadIdx.x + s]);
    __syncthreads();
  }
  if (threadIdx.x == 0) out[0] = red[0];
}

// int8 fake-quant levels are exactly representable in f16 (scale applied post-GEMM)
__global__ void k_quant_w(const float* __restrict__ w, const float* __restrict__ wamax,
                          _Float16* __restrict__ wq, int n) {
  int i = blockIdx.x * blockDim.x + threadIdx.x;
  if (i >= n) return;
  float s = wamax[0] / 127.f + 1e-12f;
  float q = fminf(fmaxf(rintf(w[i] / s), -128.f), 127.f);
  wq[i] = (_Float16)q;
}

// fold ep-projection + att_edge reduction into a tiny [2,32] matrix:
// v[h,d] = sum_c w_edge[h*64+c, d] * att_edge[h*64+c]
__global__ void k_vedge(const float* __restrict__ w_edge, const float* __restrict__ att_edge,
                        float* __restrict__ vedge) {
  int t = threadIdx.x;           // 0..63
  int h = t >> 5, d = t & 31;
  float s = 0.f;
  for (int c = 0; c < 64; ++c)
    s += w_edge[(h * 64 + c) * 32 + d] * att_edge[h * 64 + c];
  vedge[h * 32 + d] = s;
}

// ---------- node GEMM: xp = x @ wq.T * scale, via v_wmma_f32_16x16x32_f16 ----------
// block = 256 = 8 waves; each wave owns a 16x16 D-tile (16 rows x 16 cols), K=128 in 4 steps.
__global__ void __launch_bounds__(256)
k_gemm_wmma(const float* __restrict__ x, const _Float16* __restrict__ wq,
            const float* __restrict__ wamax, float* __restrict__ xp, int N) {
  const int lane = threadIdx.x & 31;
  const int wave = threadIdx.x >> 5;      // column tile 0..7
  const int half = lane >> 4;             // lane group 0/1 (K split into contiguous 16-k halves, same for A and B)
  const int lm   = lane & 15;
  const int row0 = blockIdx.x * 16;
  int row = row0 + lm;
  if (row >= N) row = N - 1;              // safe clamp; stores are guarded
  const int col = wave * 16 + lm;         // output channel (B column)

  v8f acc = {};
#pragma unroll
  for (int ks = 0; ks < 4; ++ks) {
    // A (16x32 f16): lane<16 holds K = [ka..ka+15]; lane>=16 holds K 16..31 (contiguous, matches B)
    const int ka = ks * 32 + half * 16;
    float af[16];
    *(float4*)(af + 0)  = *(const float4*)(x + row * 128 + ka + 0);
    *(float4*)(af + 4)  = *(const float4*)(x + row * 128 + ka + 4);
    *(float4*)(af + 8)  = *(const float4*)(x + row * 128 + ka + 8);
    *(float4*)(af + 12) = *(const float4*)(x + row * 128 + ka + 12);
    v16h a;
#pragma unroll
    for (int i = 0; i < 16; ++i) a[i] = (_Float16)af[i];

    // B (32x16 f16): lane-half selects K 0..15 / 16..31 for column `col`; contiguous in wq[col][k]
    v16h b = *(const v16h*)(wq + col * 128 + ks * 32 + half * 16);

    acc = __builtin_amdgcn_wmma_f32_16x16x32_f16(false, a, false, b, (short)0, acc,
                                                 false, false);
  }
  const float s = wamax[0] / 127.f + 1e-12f;
#pragma unroll
  for (int r = 0; r < 8; ++r) {
    int m = row0 + r + half * 8;          // D layout: VGPR r -> row r (lanes 0-15) / r+8 (16-31)
    if (m < N) xp[m * 128 + wave * 16 + lm] = acc[r] * s;
  }
}

// ---------- per-node attention logits ----------
__global__ void k_node_logits(const float* __restrict__ xp,
                              const float* __restrict__ att_src, const float* __restrict__ att_dst,
                              float* __restrict__ a_src, float* __restrict__ a_dst, int N) {
  int i = blockIdx.x * blockDim.x + threadIdx.x;
  if (i >= N * 2) return;
  int n = i >> 1, h = i & 1;
  const float* xr = xp + n * 128 + h * 64;
  const float* as = att_src + h * 64;
  const float* ad = att_dst + h * 64;
  float s0 = 0.f, s1 = 0.f;
#pragma unroll
  for (int c = 0; c < 64; c += 4) {
    float4 v = *(const float4*)(xr + c);
    float4 a = *(const float4*)(as + c);
    float4 b = *(const float4*)(ad + c);
    s0 += v.x * a.x + v.y * a.y + v.z * a.z + v.w * a.w;
    s1 += v.x * b.x + v.y * b.y + v.z * b.z + v.w * b.w;
  }
  a_src[i] = s0;
  a_dst[i] = s1;
}

// ---------- per-edge alpha + leaky-relu + segment max ----------
__global__ void k_edge_alpha(const float* __restrict__ ea, const int* __restrict__ ei,
                             const float* __restrict__ vedge,
                             const float* __restrict__ a_src, const float* __restrict__ a_dst,
                             float* __restrict__ alpha, float* __restrict__ amax, int E) {
  int e = blockIdx.x * blockDim.x + threadIdx.x;
  if (e >= E) return;
  int src = ei[e], dst = ei[E + e];
  float s0 = 0.f, s1 = 0.f;
#pragma unroll
  for (int d = 0; d < 32; d += 4) {
    float4 v  = *(const float4*)(ea + e * 32 + d);
    float4 v0 = *(const float4*)(vedge + d);
    float4 v1 = *(const float4*)(vedge + 32 + d);
    s0 += v.x * v0.x + v.y * v0.y + v.z * v0.z + v.w * v0.w;
    s1 += v.x * v1.x + v.y * v1.y + v.z * v1.z + v.w * v1.w;
  }
  float al0 = a_src[src * 2 + 0] + a_dst[dst * 2 + 0] + s0;
  float al1 = a_src[src * 2 + 1] + a_dst[dst * 2 + 1] + s1;
  al0 = al0 >= 0.f ? al0 : 0.2f * al0;
  al1 = al1 >= 0.f ? al1 : 0.2f * al1;
  alpha[e * 2 + 0] = al0;
  alpha[e * 2 + 1] = al1;
  atomicMaxFloat(amax + dst * 2 + 0, al0);
  atomicMaxFloat(amax + dst * 2 + 1, al1);
}

// ---------- exp + segment sum (alpha buffer overwritten with ex) ----------
__global__ void k_edge_exp(float* __restrict__ alpha, const int* __restrict__ ei,
                           const float* __restrict__ amax, float* __restrict__ denom, int E) {
  int i = blockIdx.x * blockDim.x + threadIdx.x;
  if (i >= E * 2) return;
  int e = i >> 1, h = i & 1;
  int dst = ei[E + e];
  float ex = expf(alpha[i] - amax[dst * 2 + h]);
  alpha[i] = ex;
  atomicAdd(denom + dst * 2 + h, ex);
}

// ---------- weighted message scatter: one wave per edge, lane = 4 channels ----------
__global__ void __launch_bounds__(256)
k_scatter(const float* __restrict__ xp, const float* __restrict__ ex,
          const float* __restrict__ denom, const int* __restrict__ ei,
          float* __restrict__ oacc, int E) {
  int gid = blockIdx.x * blockDim.x + threadIdx.x;
  int e = gid >> 5, lane = gid & 31;
  if (e >= E) return;
  int src = ei[e], dst = ei[E + e];
  int h = lane >> 4;   // channels 0-63 -> head 0, 64-127 -> head 1
  float coef = ex[e * 2 + h] / (denom[dst * 2 + h] + 1e-16f);
  float4 m = *(const float4*)(xp + src * 128 + lane * 4);
  float* o = oacc + dst * 128 + lane * 4;
  atomicAdd(o + 0, m.x * coef);
  atomicAdd(o + 1, m.y * coef);
  atomicAdd(o + 2, m.z * coef);
  atomicAdd(o + 3, m.w * coef);
}

// ---------- head mean + bias + global absmax ----------
__global__ void k_final1(const float* __restrict__ acc, const float* __restrict__ bias,
                         float* __restrict__ out, float* __restrict__ oamax, int N) {
  __shared__ float red[256];
  int i = blockIdx.x * blockDim.x + threadIdx.x;
  float av = 0.f;
  if (i < N * 64) {
    int n = i >> 6, c = i & 63;
    float o = 0.5f * (acc[n * 128 + c] + acc[n * 128 + 64 + c]) + bias[c];
    out[i] = o;
    av = fabsf(o);
  }
  red[threadIdx.x] = av;
  __syncthreads();
  for (int s = 128; s > 0; s >>= 1) {
    if (threadIdx.x < s) red[threadIdx.x] = fmaxf(red[threadIdx.x], red[threadIdx.x + s]);
    __syncthreads();
  }
  if (threadIdx.x == 0) atomicMax((int*)oamax, __float_as_int(red[0])); // values >= 0
}

// ---------- output 8-bit fake-quant (in place on d_out) ----------
__global__ void k_final2(const float* __restrict__ oamax, float* __restrict__ out, int n) {
  int i = blockIdx.x * blockDim.x + threadIdx.x;
  if (i >= n) return;
  float s = oamax[0] / 127.f + 1e-12f;
  float q = fminf(fmaxf(rintf(out[i] / s), -128.f), 127.f);
  out[i] = q * s;
}

extern "C" void kernel_launch(void* const* d_in, const int* in_sizes, int n_in,
                              void* d_out, int out_size, void* d_ws, size_t ws_size,
                              hipStream_t stream) {
  const float* x        = (const float*)d_in[0];
  const int*   ei       = (const int*)d_in[1];
  const float* ea       = (const float*)d_in[2];
  const float* w_lin    = (const float*)d_in[3];
  const float* w_edge   = (const float*)d_in[4];
  const float* att_src  = (const float*)d_in[5];
  const float* att_dst  = (const float*)d_in[6];
  const float* att_edge = (const float*)d_in[7];
  const float* bias     = (const float*)d_in[8];
  float* out = (float*)d_out;

  const int N = in_sizes[0] / 128;   // 100000
  const int E = in_sizes[2] / 32;    // 1600000
  (void)n_in; (void)out_size; (void)ws_size;

  char* ws = (char*)d_ws;
  size_t off = 0;
  auto alloc = [&](size_t bytes) {
    void* p = ws + off;
    off = (off + bytes + 255) & ~(size_t)255;
    return p;
  };
  float*    wamax = (float*)alloc(4);
  float*    oamax = (float*)alloc(4);
  float*    vedge = (float*)alloc(2 * 32 * sizeof(float));
  _Float16* wq    = (_Float16*)alloc(128 * 128 * sizeof(_Float16));
  float*    xp    = (float*)alloc((size_t)N * 128 * sizeof(float));
  float*    a_src = (float*)alloc((size_t)N * 2 * sizeof(float));
  float*    a_dst = (float*)alloc((size_t)N * 2 * sizeof(float));
  float*    alpha = (float*)alloc((size_t)E * 2 * sizeof(float));
  float*    amax  = (float*)alloc((size_t)N * 2 * sizeof(float));
  float*    denom = (float*)alloc((size_t)N * 2 * sizeof(float));
  float*    oacc  = (float*)alloc((size_t)N * 128 * sizeof(float));

  // prep
  k_absmax_w<<<1, 256, 0, stream>>>(w_lin, wamax, 128 * 128);
  k_quant_w<<<(16384 + 255) / 256, 256, 0, stream>>>(w_lin, wamax, wq, 16384);
  k_vedge<<<1, 64, 0, stream>>>(w_edge, att_edge, vedge);

  // re-init accumulators every call (graph replay determinism)
  k_fill<<<(N * 2 + 255) / 256, 256, 0, stream>>>(amax, -INFINITY, N * 2);
  k_fill<<<(N * 2 + 255) / 256, 256, 0, stream>>>(denom, 0.f, N * 2);
  k_fill<<<(N * 128 + 255) / 256, 256, 0, stream>>>(oacc, 0.f, N * 128);
  k_fill<<<1, 256, 0, stream>>>(oamax, 0.f, 1);

  // fused pipeline
  k_gemm_wmma<<<(N + 15) / 16, 256, 0, stream>>>(x, wq, wamax, xp, N);
  k_node_logits<<<(N * 2 + 255) / 256, 256, 0, stream>>>(xp, att_src, att_dst, a_src, a_dst, N);
  k_edge_alpha<<<(E + 255) / 256, 256, 0, stream>>>(ea, ei, vedge, a_src, a_dst, alpha, amax, E);
  k_edge_exp<<<(E * 2 + 255) / 256, 256, 0, stream>>>(alpha, ei, amax, denom, E);
  k_scatter<<<(int)(((size_t)E * 32 + 255) / 256), 256, 0, stream>>>(xp, alpha, denom, ei, oacc, E);
  k_final1<<<(N * 64 + 255) / 256, 256, 0, stream>>>(oacc, bias, out, oamax, N);
  k_final2<<<(N * 64 + 255) / 256, 256, 0, stream>>>(oamax, out, N * 64);
}
